// Goggle_62981400429116
// MI455X (gfx1250) — matrix operand order repaired and checked by reading, not compiled, not run on hardware
//
#include <hip/hip_runtime.h>
#include <hip/hip_fp16.h>
#include <math.h>

typedef __attribute__((ext_vector_type(16))) _Float16 v16h;
typedef __attribute__((ext_vector_type(8)))  _Float16 v8h;
typedef __attribute__((ext_vector_type(4)))  _Float16 v4h;
typedef __attribute__((ext_vector_type(8)))  float    v8f;
typedef __attribute__((ext_vector_type(4)))  int      v4i;

#define B_ROWS 8192
#define D      128
#define TM     32      // batch rows per workgroup -> 256 workgroups
#define RT     (TM/16) // row tiles per workgroup
#define LDH1   264     // LDS leading dim for h1 (256 + 8 pad halfs)
#define LDA    136     // LDS leading dim for 128-wide activations

typedef __attribute__((address_space(1))) v4i* as1_v4p;  // global int4*
typedef __attribute__((address_space(3))) v4i* as3_v4p;  // LDS int4*

#if defined(__gfx1250__) && __has_builtin(__builtin_amdgcn_global_load_async_to_lds_b128) && __has_builtin(__builtin_amdgcn_s_wait_asynccnt)
#define USE_ASYNC_LDS 1
#else
#define USE_ASYNC_LDS 0
#endif

// Hardware TANH trans-op when available; else 4-op exp2-based form with
// correct +-1 saturation (exp overflow -> inf -> 2/inf = 0).
__device__ __forceinline__ float fast_tanh(float x) {
#if __has_builtin(__builtin_amdgcn_tanhf)
  return __builtin_amdgcn_tanhf(x);
#elif __has_builtin(__builtin_amdgcn_tanh_f32)
  return __builtin_amdgcn_tanh_f32(x);
#else
  float e = __expf(2.f * x);
  return 1.f - 2.f / (e + 1.f);
#endif
}

// ---------------------------------------------------------------------------
// Prep kernel 1: adjacency (sigmoid + self loops + threshold), symmetric norm
// a_hat, and fused decoder vectors  wfg1 = W_feat·W_g1, b_g1, W_g2, b_g2.
// ---------------------------------------------------------------------------
__global__ void goggle_graph_prep(const float* __restrict__ logits,
                                  const int*   __restrict__ itp,
                                  const float* __restrict__ W_feat,
                                  const float* __restrict__ W_g1,
                                  const float* __restrict__ b_g1,
                                  const float* __restrict__ W_g2,
                                  const float* __restrict__ b_g2,
                                  float* __restrict__ adj_out,
                                  float* __restrict__ adj_ws,
                                  float* __restrict__ ahat_ws,
                                  float* __restrict__ par_ws) {
  __shared__ float rowsum[D], colsum[D];
  const int t = threadIdx.x;        // one row per thread, 128 threads
  const int it = itp[0];
  float rs = 0.f;
  for (int j = 0; j < D; ++j) {
    float g = 1.f / (1.f + __expf(-logits[t * D + j]));
    if (t == j)                    g = 1.f;
    else if (it > 50 && g < 0.1f)  g = 0.f;
    adj_ws[t * D + j] = g;
    adj_out[t * D + j] = g;
    rs += (g > 0.f) ? 1.f : 0.f;
  }
  rowsum[t] = fmaxf(rs, 1.f);
  __syncthreads();
  float cs = 0.f;
  for (int i = 0; i < D; ++i) cs += (adj_ws[i * D + t] > 0.f) ? 1.f : 0.f;
  colsum[t] = fmaxf(cs, 1.f);
  __syncthreads();
  const float ri = rsqrtf(rowsum[t]);
  for (int j = 0; j < D; ++j)
    ahat_ws[t * D + j] = adj_ws[t * D + j] * ri * rsqrtf(colsum[j]);
  if (t < 32) {
    float s = 0.f;
    for (int f = 0; f < 64; ++f) s += W_feat[f] * W_g1[f * 32 + t];
    par_ws[t]      = s;          // wfg1[c]
    par_ws[32 + t] = b_g1[t];
    par_ws[64 + t] = W_g2[t];
  }
  if (t == 0) par_ws[96] = b_g2[0];
}

// ---------------------------------------------------------------------------
// Prep kernel 2: pack an f32 row-major (K x N) matrix into f16 WMMA B-matrix
// fragments. CDNA5 f16 B 32x16 layout: n = lane&15, K = e + 8*(lane>>4) + 8*(e>=8).
// Fragment f = ct*(K/32) + kt; element address = (f*32 + lane)*16 + e.
// ---------------------------------------------------------------------------
__global__ void goggle_prepack(const float* __restrict__ W,
                               _Float16* __restrict__ dst, int K, int N) {
  const int gid = blockIdx.x * 256 + threadIdx.x;
  const int total = (K / 32) * (N / 16) * 32;
  if (gid >= total) return;
  const int lane = gid & 31;
  const int f    = gid >> 5;
  const int KT   = K / 32;
  const int ct   = f / KT;
  const int kt   = f - ct * KT;
  const int n     = ct * 16 + (lane & 15);
  const int kbase = kt * 32 + 8 * (lane >> 4);
  _Float16* o = dst + (size_t)gid * 16;
#pragma unroll
  for (int e = 0; e < 16; ++e)
    o[e] = (_Float16)W[(size_t)(kbase + e + (e >= 8 ? 8 : 0)) * N + n];
}

// ---------------------------------------------------------------------------
// Prep kernel 3: convert x (8192x128 f32) -> f16 once, so the main kernel can
// async-DMA rows straight into LDS in fragment-ready element size.
// ---------------------------------------------------------------------------
__global__ void goggle_xcvt(const float* __restrict__ x,
                            _Float16* __restrict__ xh) {
  const size_t gid = (size_t)blockIdx.x * 256 + threadIdx.x;  // one per 4 elems
  float4 v = ((const float4*)x)[gid];
  v4h o = {(_Float16)v.x, (_Float16)v.y, (_Float16)v.z, (_Float16)v.w};
  *(v4h*)(xh + gid * 4) = o;
}

// ---------------------------------------------------------------------------
// Fragment loaders.
// A 16x32 f16 layout: m = lane&15, K(e) = e + 8*(lane>>4) + 8*(e>=8)
//   -> two contiguous 8-half (16 B) LDS loads.
// ---------------------------------------------------------------------------
__device__ __forceinline__ v16h load_afrag(const _Float16* act, int ld,
                                           int rt, int kt, int lane) {
  const int m  = rt * 16 + (lane & 15);
  const int hi = lane >> 4;
  const _Float16* p = act + m * ld + kt * 32 + 8 * hi;
  union { v16h v; v8h h[2]; } u;
  u.h[0] = *(const v8h*)(p);
  u.h[1] = *(const v8h*)(p + 16);
  return u.v;
}

__device__ __forceinline__ v16h load_bfrag(const _Float16* wp, int kTiles,
                                           int ct, int kt, int lane) {
  return *(const v16h*)(wp + (((size_t)(ct * kTiles + kt)) * 32 + lane) * 16);
}

// Generic GEMM stage: 8 waves own CTW column tiles each; B fragments hoisted
// for all (ct,kt); A fragments loaded once per (rt,kt) and reused across cts.
template <int KT, int CTW, typename Epi>
__device__ __forceinline__ void gemm_stage(const _Float16* act, int ald,
                                           const _Float16* wp,
                                           int lane, int wave, Epi epi) {
  v16h bf[CTW][KT];
#pragma unroll
  for (int j = 0; j < CTW; ++j)
#pragma unroll
    for (int kt = 0; kt < KT; ++kt)
      bf[j][kt] = load_bfrag(wp, KT, wave * CTW + j, kt, lane);
#pragma unroll
  for (int rt = 0; rt < RT; ++rt) {
    v16h af[KT];
#pragma unroll
    for (int kt = 0; kt < KT; ++kt) af[kt] = load_afrag(act, ald, rt, kt, lane);
#pragma unroll
    for (int j = 0; j < CTW; ++j) {
      v8f acc = {};
#pragma unroll
      for (int kt = 0; kt < KT; ++kt)
        acc = __builtin_amdgcn_wmma_f32_16x16x32_f16(
            false, af[kt], false, bf[j][kt], (short)0, acc, false, false);
      epi(rt, wave * CTW + j, acc);
    }
  }
}

// ---------------------------------------------------------------------------
// Fully-fused main kernel (per 32-row tile):
//   h1=relu(x W1+b1); h2=relu(h1 W2+b2); mu,lv heads; z reparam;
//   s=z a_hat; t=sum_c wg2*tanh(s*wfg1+bg1); x_hat=t a_hat + bg2.
// All intermediates stay in LDS (f16); only x/eps in, x_hat/mu/lv out.
// ---------------------------------------------------------------------------
__global__ __launch_bounds__(256)
void goggle_main(const _Float16* __restrict__ xh, const float* __restrict__ eps,
                 const _Float16* __restrict__ pW1,  const float* __restrict__ b1,
                 const _Float16* __restrict__ pW2,  const float* __restrict__ b2,
                 const _Float16* __restrict__ pWmu, const float* __restrict__ bmu,
                 const _Float16* __restrict__ pWlv, const float* __restrict__ blv,
                 const _Float16* __restrict__ pA,
                 const float* __restrict__ par_ws,
                 float* __restrict__ xhat, float* __restrict__ muOut,
                 float* __restrict__ lvOut) {
  __shared__ __align__(16) _Float16 sh1[TM * LDH1];  // h1 (32 x 256)
  __shared__ __align__(16) _Float16 sa [TM * LDA];   // x -> h2 -> t
  __shared__ __align__(16) _Float16 sz [TM * LDA];   // z
  __shared__ float par[128];

  const int tid  = threadIdx.x;
  const int lane = tid & 31;
  const int wave = tid >> 5;
  const int hi   = lane >> 4;
  const int nn   = lane & 15;
  const size_t rowBase = (size_t)blockIdx.x * TM;

  if (tid < 97) par[tid] = par_ws[tid];

  // x tile (f16) -> LDS: 16-byte chunks, async DMA when available
#pragma unroll
  for (int i = tid; i < TM * 16; i += 256) {   // TM*16 chunks of 8 halfs
    const int r = i >> 4, c8 = i & 15;
    const _Float16* gsrc = xh + (rowBase + r) * D + c8 * 8;
    _Float16* ldst = sa + r * LDA + c8 * 8;
#if USE_ASYNC_LDS
    __builtin_amdgcn_global_load_async_to_lds_b128((as1_v4p)gsrc, (as3_v4p)ldst,
                                                   0, 0);
#else
    *(v8h*)ldst = *(const v8h*)gsrc;
#endif
  }
#if USE_ASYNC_LDS
  __builtin_amdgcn_s_wait_asynccnt(0);
#endif
  __syncthreads();

  // Stage A: h1 = relu(x @ W1 + b1)   (K=128 -> 4 ktiles, N=256 -> 2 ct/wave)
  gemm_stage<4, 2>(sa, LDA, pW1, lane, wave, [&](int rt, int ct, v8f acc) {
    const float b = b1[ct * 16 + nn];
#pragma unroll
    for (int e = 0; e < 8; ++e) {
      float v = acc[e] + b;
      v = v > 0.f ? v : 0.f;
      sh1[(rt * 16 + e + 8 * hi) * LDH1 + ct * 16 + nn] = (_Float16)v;
    }
  });
  __syncthreads();

  // Stage B: h2 = relu(h1 @ W2 + b2)  (K=256 -> 8 ktiles, N=128 -> 1 ct/wave)
  gemm_stage<8, 1>(sh1, LDH1, pW2, lane, wave, [&](int rt, int ct, v8f acc) {
    const float b = b2[ct * 16 + nn];
#pragma unroll
    for (int e = 0; e < 8; ++e) {
      float v = acc[e] + b;
      v = v > 0.f ? v : 0.f;
      sa[(rt * 16 + e + 8 * hi) * LDA + ct * 16 + nn] = (_Float16)v;
    }
  });
  __syncthreads();

  // Stage C+D: mu & lv heads share A fragments; reparameterize z in epilogue.
  {
    const int ct = wave;  // N=128 -> 1 column tile per wave
    v16h bM[4], bL[4];
#pragma unroll
    for (int kt = 0; kt < 4; ++kt) {
      bM[kt] = load_bfrag(pWmu, 4, ct, kt, lane);
      bL[kt] = load_bfrag(pWlv, 4, ct, kt, lane);
    }
    const float bm = bmu[ct * 16 + nn];
    const float bl = blv[ct * 16 + nn];
#pragma unroll
    for (int rt = 0; rt < RT; ++rt) {
      v16h af[4];
#pragma unroll
      for (int kt = 0; kt < 4; ++kt) af[kt] = load_afrag(sa, LDA, rt, kt, lane);
      v8f aM = {}, aL = {};
#pragma unroll
      for (int kt = 0; kt < 4; ++kt) {
        aM = __builtin_amdgcn_wmma_f32_16x16x32_f16(false, af[kt], false, bM[kt],
                                                    (short)0, aM, false, false);
        aL = __builtin_amdgcn_wmma_f32_16x16x32_f16(false, af[kt], false, bL[kt],
                                                    (short)0, aL, false, false);
      }
#pragma unroll
      for (int e = 0; e < 8; ++e) {
        const int row = rt * 16 + e + 8 * hi;
        const int col = ct * 16 + nn;
        const size_t g = (rowBase + row) * (size_t)D + col;
        const float m = aM[e] + bm;
        const float l = aL[e] + bl;
        muOut[g] = m;
        lvOut[g] = l;
        const float z = m + eps[g] * __expf(0.5f * l);
        sz[row * LDA + col] = (_Float16)z;
      }
    }
  }
  __syncthreads();

  // Stage E+F: s = z @ a_hat, then t = sum_c wg2[c]*tanh(s*wfg1[c]+bg1[c])
  gemm_stage<4, 1>(sz, LDA, pA, lane, wave, [&](int rt, int ct, v8f acc) {
#pragma unroll
    for (int e = 0; e < 8; ++e) {
      const float s = acc[e];
      float t = 0.f;
#pragma unroll
      for (int c = 0; c < 32; ++c)
        t += par[64 + c] * fast_tanh(fmaf(s, par[c], par[32 + c]));
      sa[(rt * 16 + e + 8 * hi) * LDA + ct * 16 + nn] = (_Float16)t;
    }
  });
  __syncthreads();

  // Stage G: x_hat = t @ a_hat + b_g2
  const float bg2v = par[96];
  gemm_stage<4, 1>(sa, LDA, pA, lane, wave, [&](int rt, int ct, v8f acc) {
#pragma unroll
    for (int e = 0; e < 8; ++e) {
      const size_t g =
          (rowBase + rt * 16 + e + 8 * hi) * (size_t)D + ct * 16 + nn;
      xhat[g] = acc[e] + bg2v;
    }
  });
}

// ---------------------------------------------------------------------------
extern "C" void kernel_launch(void* const* d_in, const int* in_sizes, int n_in,
                              void* d_out, int out_size, void* d_ws,
                              size_t ws_size, hipStream_t stream) {
  const float* x      = (const float*)d_in[0];
  const float* eps    = (const float*)d_in[1];
  const float* W1     = (const float*)d_in[2];
  const float* b1     = (const float*)d_in[3];
  const float* W2     = (const float*)d_in[4];
  const float* b2     = (const float*)d_in[5];
  const float* Wmu    = (const float*)d_in[6];
  const float* bmu    = (const float*)d_in[7];
  const float* Wlv    = (const float*)d_in[8];
  const float* blv    = (const float*)d_in[9];
  const float* logits = (const float*)d_in[10];
  const float* Wf     = (const float*)d_in[11];
  const float* Wg1    = (const float*)d_in[12];
  const float* bg1    = (const float*)d_in[13];
  const float* Wg2    = (const float*)d_in[14];
  const float* bg2    = (const float*)d_in[15];
  const int*   itp    = (const int*)d_in[16];

  // output tuple flat: x_hat (8192*128), adj (128*128), mu, logvar
  float* out  = (float*)d_out;
  float* xhat = out;
  float* adjO = out + (size_t)B_ROWS * D;
  float* muO  = adjO + (size_t)D * D;
  float* lvO  = muO + (size_t)B_ROWS * D;

  // workspace layout (all offsets 32-B aligned)
  char* ws = (char*)d_ws;
  float*    adj_ws  = (float*)(ws);              //  64 KB
  float*    ahat_ws = (float*)(ws + 65536);      //  64 KB
  float*    par_ws  = (float*)(ws + 131072);     //  512 B
  _Float16* pW1     = (_Float16*)(ws + 131584);  //  64 KB
  _Float16* pW2     = (_Float16*)(ws + 197120);  //  64 KB
  _Float16* pWmu    = (_Float16*)(ws + 262656);  //  32 KB
  _Float16* pWlv    = (_Float16*)(ws + 295424);  //  32 KB
  _Float16* pA      = (_Float16*)(ws + 328192);  //  32 KB
  _Float16* xh      = (_Float16*)(ws + 360960);  //   2 MB (x in f16)

  goggle_graph_prep<<<1, 128, 0, stream>>>(logits, itp, Wf, Wg1, bg1, Wg2, bg2,
                                           adjO, adj_ws, ahat_ws, par_ws);
  goggle_xcvt<<<(B_ROWS * D / 4) / 256, 256, 0, stream>>>(x, xh);
  goggle_prepack<<<8, 256, 0, stream>>>(W1, pW1, 128, 256);
  goggle_prepack<<<8, 256, 0, stream>>>(W2, pW2, 256, 128);
  goggle_prepack<<<4, 256, 0, stream>>>(Wmu, pWmu, 128, 128);
  goggle_prepack<<<4, 256, 0, stream>>>(Wlv, pWlv, 128, 128);
  goggle_prepack<<<4, 256, 0, stream>>>(ahat_ws, pA, 128, 128);

  goggle_main<<<B_ROWS / TM, 256, 0, stream>>>(xh, eps, pW1, b1, pW2, b2, pWmu,
                                               bmu, pWlv, blv, pA, par_ws,
                                               xhat, muO, lvO);
}
